// SelfAttention_33354716021182
// MI455X (gfx1250) — matrix operand order repaired and checked by reading, not compiled, not run on hardware
//
#include <hip/hip_runtime.h>
#include <math.h>
#include <stdint.h>

typedef __bf16 v16bf __attribute__((ext_vector_type(16)));
typedef __bf16 v8bf  __attribute__((ext_vector_type(8)));
typedef float  v8f   __attribute__((ext_vector_type(8)));

#define WMMA_BF16(a, b, c) \
  __builtin_amdgcn_wmma_f32_16x16x32_bf16(false, (a), false, (b), (short)0, (c), false, false)

constexpr int kB = 16, kN = 1029, kC = 1024, kH = 16, kD = 64;
constexpr int kMreal  = kB * kN;        // 16464 real rows
constexpr int kMpad   = 16512;          // 129 * 128
constexpr int kNPAD   = 1088;           // 17*64 seq padding
constexpr int kPrefix = 5;
constexpr int kKT     = (kN + 63) / 64; // 17 key tiles of 64

// LDS row strides (bf16 elements): 16 consecutive-row b128 readers hit 16
// distinct banks, 16B alignment kept.
constexpr int kLdA = 40;  // 32-col tiles (80B  = 20 banks)
constexpr int kLdK = 72;  // 64-col tiles (144B = 36 banks)

__device__ __forceinline__ float  bf2f(__bf16 v) { return (float)v; }
__device__ __forceinline__ __bf16 f2bf(float f)  { return (__bf16)f; }

__device__ __forceinline__ v8f vzero8() {
  v8f z;
#pragma unroll
  for (int i = 0; i < 8; ++i) z[i] = 0.0f;
  return z;
}

// Async global->LDS copy of 16 bytes per lane (ASYNCcnt-tracked path).
__device__ __forceinline__ void async_b128(const void* gptr, const void* lptr) {
  unsigned lds = (unsigned)(uintptr_t)lptr;               // low 32b = LDS offset
  unsigned long long g = (unsigned long long)(uintptr_t)gptr;
  asm volatile("global_load_async_to_lds_b128 %0, %1, off"
               :: "v"(lds), "v"(g) : "memory");
}
__device__ __forceinline__ void wait_async_le6() {
  asm volatile("s_wait_asynccnt 0x6" ::: "memory");
}
__device__ __forceinline__ void wait_async_le8() {
  asm volatile("s_wait_asynccnt 0x8" ::: "memory");
}
__device__ __forceinline__ void wait_async_0() {
  asm volatile("s_wait_asynccnt 0x0" ::: "memory");
}

// A-fragment (16x32 bf16, MxK): lane l<16: row m=l, K={0..7,16..23};
// lane l>=16: row m=l-16, K={8..15,24..31}.  (ISA 7.12.2)
__device__ __forceinline__ v16bf frag_a(const __bf16* base, int ld,
                                        int row0, int k0, int lane) {
  const int m  = row0 + (lane & 15);
  const int kh = (lane >> 4) << 3;
  const __bf16* p = base + (size_t)m * ld + k0 + kh;
  v8bf lo = *(const v8bf*)p;
  v8bf hi = *(const v8bf*)(p + 16);
  return __builtin_shufflevector(lo, hi, 0, 1, 2, 3, 4, 5, 6, 7,
                                 8, 9, 10, 11, 12, 13, 14, 15);
}

// B-fragment (32x16 bf16, KxN) from a [N, K] row-major array (B^T):
// lane l<16: column n=l, K={0..7,16..23}; lane l>=16: n=l-16, K={8..15,24..31}.
__device__ __forceinline__ v16bf frag_b(const __bf16* base, int ld,
                                        int col0, int k0, int lane) {
  const int n  = col0 + (lane & 15);
  const int kb = (lane >> 4) << 3;
  const __bf16* p = base + (size_t)n * ld + k0 + kb;
  v8bf lo = *(const v8bf*)p;
  v8bf hi = *(const v8bf*)(p + 16);
  return __builtin_shufflevector(lo, hi, 0, 1, 2, 3, 4, 5, 6, 7,
                                 8, 9, 10, 11, 12, 13, 14, 15);
}

// ---------------------------------------------------------------- pack kernels
__global__ void pack_pad_kernel(const float* __restrict__ in, __bf16* __restrict__ out,
                                size_t nvalid, size_t ntot) {
  size_t i = (size_t)blockIdx.x * blockDim.x + threadIdx.x;
  if (i < ntot) out[i] = (i < nvalid) ? f2bf(in[i]) : (__bf16)0.0f;
}

__global__ void pack_kernel(const float* __restrict__ in, __bf16* __restrict__ out,
                            size_t n) {
  size_t i = (size_t)blockIdx.x * blockDim.x + threadIdx.x;
  if (i < n) out[i] = f2bf(in[i]);
}

// --------------------------------------------- WMMA GEMM, LDS double-buffered
// A: [Mpad, K] bf16.  Bw: [Nout, K] bf16 (= B^T).  Block tile 128x256,
// 8 waves (2M x 4N), wave tile 64x64.  Per K-step of 32: stage A(128x32) +
// B(256x32) via async-to-LDS (6 b128/thread), 16 WMMAs per wave from LDS.
template <bool OUT_F32>
__global__ __launch_bounds__(256) void gemm_bias_kernel(
    const __bf16* __restrict__ A, const __bf16* __restrict__ Bw,
    const float* __restrict__ bias, void* __restrict__ Yv,
    int K, int Nout, int MrealStore) {
  __shared__ __align__(16) __bf16 tA[2][128][kLdA];
  __shared__ __align__(16) __bf16 tB[2][256][kLdA];

  const int tid  = threadIdx.x;
  const int lane = tid & 31;
  const int w    = tid >> 5;
  const int blockM = blockIdx.y * 128;
  const int blockN = blockIdx.x * 256;
  const int wm = (w & 1) * 64;
  const int wn = (w >> 1) * 64;

  auto stage = [&](int buf, int k0) {
#pragma unroll
    for (int s = 0; s < 2; ++s) {            // A: 512 chunks
      const int c = tid + s * 256;
      const int row = c >> 2, sub = (c & 3) << 3;
      async_b128(A + (size_t)(blockM + row) * K + k0 + sub, &tA[buf][row][sub]);
    }
#pragma unroll
    for (int s = 0; s < 4; ++s) {            // B: 1024 chunks
      const int c = tid + s * 256;
      const int row = c >> 2, sub = (c & 3) << 3;
      async_b128(Bw + (size_t)(blockN + row) * K + k0 + sub, &tB[buf][row][sub]);
    }
  };

  v8f acc[4][4];
#pragma unroll
  for (int i = 0; i < 4; ++i)
#pragma unroll
    for (int j = 0; j < 4; ++j) acc[i][j] = vzero8();

  const int nk = K >> 5;
  stage(0, 0);
  for (int kt = 0; kt < nk; ++kt) {
    if (kt + 1 < nk) {
      stage((kt + 1) & 1, (kt + 1) << 5);
      wait_async_le6();                      // previous buffer's 6 copies done
    } else {
      wait_async_0();
    }
    __syncthreads();                         // tiles visible to all waves

    const __bf16* tAb = &tA[kt & 1][0][0];
    const __bf16* tBb = &tB[kt & 1][0][0];
    v16bf bfr[4];
#pragma unroll
    for (int j = 0; j < 4; ++j) bfr[j] = frag_b(tBb, kLdA, wn + 16 * j, 0, lane);
#pragma unroll
    for (int i = 0; i < 4; ++i) {
      v16bf af = frag_a(tAb, kLdA, wm + 16 * i, 0, lane);
#pragma unroll
      for (int j = 0; j < 4; ++j) acc[i][j] = WMMA_BF16(af, bfr[j], acc[i][j]);
    }
    __syncthreads();                         // safe to overwrite this buffer
  }

  const int hi = lane >> 4, cl = lane & 15;
#pragma unroll
  for (int i = 0; i < 4; ++i) {
#pragma unroll
    for (int j = 0; j < 4; ++j) {
      const int col = blockN + wn + 16 * j + cl;
      const float bv = bias[col];
#pragma unroll
      for (int r = 0; r < 8; ++r) {
        const int row = blockM + wm + 16 * i + r + 8 * hi;
        const float v = acc[i][j][r] + bv;
        if (OUT_F32) {
          if (row < MrealStore) ((float*)Yv)[(size_t)row * Nout + col] = v;
        } else {
          ((__bf16*)Yv)[(size_t)row * Nout + col] = f2bf(v);
        }
      }
    }
  }
}

// ------------------------------------------------- RoPE + scatter to Q/K/V layouts
__global__ void rope_scatter_kernel(const __bf16* __restrict__ qkv,
                                    const float* __restrict__ rsin,
                                    const float* __restrict__ rcos,
                                    __bf16* __restrict__ Q, __bf16* __restrict__ Kd,
                                    __bf16* __restrict__ Vt) {
  const int t = blockIdx.x * blockDim.x + threadIdx.x;
  if (t >= kB * kH * kNPAD) return;
  const int bh = t / kNPAD;
  const int n  = t % kNPAD;
  const int b  = bh >> 4;
  const int h  = bh & 15;
  const size_t qrow  = ((size_t)bh * kNPAD + n) * kD;
  const size_t vbase = (size_t)bh * kD * kNPAD;

  if (n >= kN) {
#pragma unroll 4
    for (int d = 0; d < kD; ++d) {
      Q[qrow + d]  = (__bf16)0.0f;
      Kd[qrow + d] = (__bf16)0.0f;
      Vt[vbase + (size_t)d * kNPAD + n] = (__bf16)0.0f;
    }
    return;
  }

  const __bf16* src = qkv + (size_t)(b * kN + n) * 3072;
  const int hc = h * kD;
  const bool doRope = (n >= kPrefix);
  const int ridx = n - kPrefix;

#pragma unroll 4
  for (int d = 0; d < 32; ++d) {
    float q1 = bf2f(src[hc + d]),      q2 = bf2f(src[hc + d + 32]);
    float k1 = bf2f(src[kC + hc + d]), k2 = bf2f(src[kC + hc + d + 32]);
    float oq1 = q1, oq2 = q2, ok1 = k1, ok2 = k2;
    if (doRope) {
      const float c1 = rcos[ridx * kD + d],      s1 = rsin[ridx * kD + d];
      const float c2 = rcos[ridx * kD + d + 32], s2 = rsin[ridx * kD + d + 32];
      oq1 = q1 * c1 - q2 * s1;  oq2 = q2 * c2 + q1 * s2;
      ok1 = k1 * c1 - k2 * s1;  ok2 = k2 * c2 + k1 * s2;
    }
    Q[qrow + d]       = f2bf(oq1 * 0.125f);   // fold 1/sqrt(64)
    Q[qrow + d + 32]  = f2bf(oq2 * 0.125f);
    Kd[qrow + d]      = f2bf(ok1);
    Kd[qrow + d + 32] = f2bf(ok2);
  }
#pragma unroll 4
  for (int d = 0; d < kD; ++d)
    Vt[vbase + (size_t)d * kNPAD + n] = src[2 * kC + hc + d];
}

// --------------------------------------------------------------- flash attention
// Grid (17 q-tiles, 256 bh), 128 threads = 4 waves.  64-key tiles: 16 WMMAs
// per tile against one set of cross-lane softmax reductions.  K/V^T tiles are
// staged into LDS with async-to-LDS copies shared by all 4 waves; Q fragments
// live in registers for the whole key loop.
__global__ __launch_bounds__(128) void attn_kernel(
    const __bf16* __restrict__ Q, const __bf16* __restrict__ Kd,
    const __bf16* __restrict__ Vt, __bf16* __restrict__ Obf) {
  __shared__ __align__(16) __bf16 sK[2][64][kLdK];   // [key][d]
  __shared__ __align__(16) __bf16 sV[2][64][kLdK];   // [d][key]
  __shared__ __align__(16) __bf16 Plds[4][16][kLdK]; // per-wave 16x64 P tile

  const int tid  = threadIdx.x;
  const int lane = tid & 31;
  const int w    = tid >> 5;
  const int bh   = blockIdx.y;
  const int b    = bh >> 4, h = bh & 15;
  const int q0   = blockIdx.x * 64 + w * 16;
  const int hi   = lane >> 4, cl = lane & 15;

  const __bf16* Qb = Q + (size_t)bh * kNPAD * kD;
  const __bf16* Kb = Kd + (size_t)bh * kNPAD * kD;
  const __bf16* Vb = Vt + (size_t)bh * kD * kNPAD;

  auto stage = [&](int buf, int key0) {
#pragma unroll
    for (int s = 0; s < 4; ++s) {
      const int c = tid + s * 128;                  // 512 chunks per tile
      const int kr = c >> 3, ks = (c & 7) << 3;     // 64 rows x 8 chunks
      async_b128(Kb + (size_t)(key0 + kr) * kD + ks, &sK[buf][kr][ks]);
      async_b128(Vb + (size_t)kr * kNPAD + key0 + ks, &sV[buf][kr][ks]);
    }
  };

  // Q fragments: reused for all key tiles
  v16bf aq0 = frag_a(Qb, kD, q0, 0, lane);
  v16bf aq1 = frag_a(Qb, kD, q0, 32, lane);

  v8f accO[4];
#pragma unroll
  for (int j = 0; j < 4; ++j) accO[j] = vzero8();
  float rm[8], rl[8];
#pragma unroll
  for (int r = 0; r < 8; ++r) { rm[r] = -1e30f; rl[r] = 0.0f; }

  stage(0, 0);
  for (int kt = 0; kt < kKT; ++kt) {
    const int key0 = kt * 64;
    if (kt + 1 < kKT) {
      stage((kt + 1) & 1, (kt + 1) * 64);
      wait_async_le8();
    } else {
      wait_async_0();
    }
    __syncthreads();

    const __bf16* sKb = &sK[kt & 1][0][0];
    const __bf16* sVb = &sV[kt & 1][0][0];

    // ---- S = Q K^T  (16x64 tile as four 16x16 accumulators)
    v8f s[4];
#pragma unroll
    for (int j = 0; j < 4; ++j) s[j] = vzero8();
#pragma unroll
    for (int j = 0; j < 4; ++j) s[j] = WMMA_BF16(aq0, frag_b(sKb, kLdK, 16 * j, 0, lane), s[j]);
#pragma unroll
    for (int j = 0; j < 4; ++j) s[j] = WMMA_BF16(aq1, frag_b(sKb, kLdK, 16 * j, 32, lane), s[j]);

    // ---- online softmax (row m = r + 8*hi, per 16-lane group)
    bool valid[4];
#pragma unroll
    for (int j = 0; j < 4; ++j) valid[j] = (key0 + 16 * j + cl) < kN;
#pragma unroll
    for (int r = 0; r < 8; ++r) {
      float p[4];
#pragma unroll
      for (int j = 0; j < 4; ++j) p[j] = valid[j] ? s[j][r] : -3.0e38f;
      float tm = fmaxf(fmaxf(p[0], p[1]), fmaxf(p[2], p[3]));
      tm = fmaxf(tm, __shfl_xor(tm, 1, 32));
      tm = fmaxf(tm, __shfl_xor(tm, 2, 32));
      tm = fmaxf(tm, __shfl_xor(tm, 4, 32));
      tm = fmaxf(tm, __shfl_xor(tm, 8, 32));
      const float Mn = fmaxf(rm[r], tm);
      const float alpha = __expf(rm[r] - Mn);
      rm[r] = Mn;
#pragma unroll
      for (int j = 0; j < 4; ++j) p[j] = valid[j] ? __expf(p[j] - Mn) : 0.0f;
      float rs = (p[0] + p[1]) + (p[2] + p[3]);
      rs += __shfl_xor(rs, 1, 32);
      rs += __shfl_xor(rs, 2, 32);
      rs += __shfl_xor(rs, 4, 32);
      rs += __shfl_xor(rs, 8, 32);
      rl[r] = rl[r] * alpha + rs;
#pragma unroll
      for (int j = 0; j < 4; ++j) accO[j][r] *= alpha;
#pragma unroll
      for (int j = 0; j < 4; ++j) Plds[w][r + 8 * hi][16 * j + cl] = f2bf(p[j]);
    }

    // ---- O += P V (P via LDS to reach A-fragment layout; two K-chunks)
    {
      const __bf16* pbase = &Plds[w][0][0];
      v16bf pf0 = frag_a(pbase, kLdK, 0, 0, lane);
      v16bf pf1 = frag_a(pbase, kLdK, 0, 32, lane);
#pragma unroll
      for (int j = 0; j < 4; ++j)
        accO[j] = WMMA_BF16(pf0, frag_b(sVb, kLdK, 16 * j, 0, lane), accO[j]);
#pragma unroll
      for (int j = 0; j < 4; ++j)
        accO[j] = WMMA_BF16(pf1, frag_b(sVb, kLdK, 16 * j, 32, lane), accO[j]);
    }
    __syncthreads();
  }

  // ---- normalize + store (row b*N+q, col h*64+d)
#pragma unroll
  for (int j = 0; j < 4; ++j) {
#pragma unroll
    for (int r = 0; r < 8; ++r) {
      const int qi = q0 + r + 8 * hi;
      if (qi < kN) {
        const float v = accO[j][r] / rl[r];
        Obf[((size_t)b * kN + qi) * kC + h * kD + 16 * j + cl] = f2bf(v);
      }
    }
  }
}

// ---------------------------------------------------------------------- launch
extern "C" void kernel_launch(void* const* d_in, const int* in_sizes, int n_in,
                              void* d_out, int out_size, void* d_ws, size_t ws_size,
                              hipStream_t stream) {
  const float* x     = (const float*)d_in[0];
  const float* rsin  = (const float*)d_in[1];
  const float* rcos  = (const float*)d_in[2];
  const float* Wqkv  = (const float*)d_in[3];
  const float* bqkv  = (const float*)d_in[4];
  const float* Wproj = (const float*)d_in[5];
  const float* bproj = (const float*)d_in[6];
  float* out = (float*)d_out;

  char* ws = (char*)d_ws;
  size_t off = 0;
  auto wsalloc = [&](size_t bytes) -> void* {
    void* p = ws + off;
    off += (bytes + 255) & ~(size_t)255;
    return p;
  };
  __bf16* Xbf   = (__bf16*)wsalloc((size_t)kMpad * kC * 2);   // reused as Obf
  __bf16* Wqb   = (__bf16*)wsalloc((size_t)3 * kC * kC * 2);
  __bf16* Wpb   = (__bf16*)wsalloc((size_t)kC * kC * 2);
  __bf16* qkvbf = (__bf16*)wsalloc((size_t)kMpad * 3 * kC * 2);
  __bf16* Qb    = (__bf16*)wsalloc((size_t)kB * kH * kNPAD * kD * 2);
  __bf16* Kb    = (__bf16*)wsalloc((size_t)kB * kH * kNPAD * kD * 2);
  __bf16* Vtb   = (__bf16*)wsalloc((size_t)kB * kH * kD * kNPAD * 2);
  __bf16* Obf   = Xbf;  // Xbf dead after QKV GEMM

  const size_t nx = (size_t)kMpad * kC;
  pack_pad_kernel<<<dim3((unsigned)((nx + 255) / 256)), dim3(256), 0, stream>>>(
      x, Xbf, (size_t)kMreal * kC, nx);
  pack_kernel<<<dim3((3 * kC * kC) / 256), dim3(256), 0, stream>>>(
      Wqkv, Wqb, (size_t)3 * kC * kC);
  pack_kernel<<<dim3((kC * kC) / 256), dim3(256), 0, stream>>>(
      Wproj, Wpb, (size_t)kC * kC);

  gemm_bias_kernel<false><<<dim3(3 * kC / 256, kMpad / 128), dim3(256), 0, stream>>>(
      Xbf, Wqb, bqkv, qkvbf, kC, 3 * kC, kMpad);

  const int nrt = kB * kH * kNPAD;
  rope_scatter_kernel<<<dim3((nrt + 255) / 256), dim3(256), 0, stream>>>(
      qkvbf, rsin, rcos, Qb, Kb, Vtb);

  attn_kernel<<<dim3(kNPAD / 64, kB * kH), dim3(128), 0, stream>>>(Qb, Kb, Vtb, Obf);

  gemm_bias_kernel<true><<<dim3(kC / 256, kMpad / 128), dim3(256), 0, stream>>>(
      Obf, Wpb, bproj, out, kC, kC, kMreal);
}
